// GraphPolicyNetwork_27462020891007
// MI455X (gfx1250) — matrix operand order repaired and checked by reading, not compiled, not run on hardware
//
#include <hip/hip_runtime.h>
#include <math.h>

#define N_NODES 40000
#define NV      40001      // +graph token row 0
#define NPAD    40016      // padded to multiple of 16 for WMMA tiles
#define NE      640000
#define HEADS   4
#define KDIM    128        // every GEMM in this net has K=128
#define KTILES  (KDIM / 32)

typedef __attribute__((ext_vector_type(16))) __bf16 v16bf;
typedef __attribute__((ext_vector_type(8)))  float  v8f;
typedef __attribute__((ext_vector_type(4)))  unsigned int v4u;
typedef __attribute__((ext_vector_type(4)))  int v4i;
typedef __attribute__((ext_vector_type(8)))  int v8i;

// ---------- scalar helpers ----------
__device__ __forceinline__ unsigned short f32_to_bf16(float f) {
  unsigned u = __float_as_uint(f);
  unsigned r = u + 0x7fffu + ((u >> 16) & 1u);
  return (unsigned short)(r >> 16);
}
__device__ __forceinline__ float bf16_to_f32(unsigned short h) {
  return __uint_as_float(((unsigned)h) << 16);
}
// monotonic float<->uint mapping for atomicMax-based segment_max
__device__ __forceinline__ unsigned ford(float f) {
  unsigned u = __float_as_uint(f);
  return (u & 0x80000000u) ? ~u : (u | 0x80000000u);
}
__device__ __forceinline__ float funord(unsigned o) {
  unsigned u = (o & 0x80000000u) ? (o & 0x7fffffffu) : ~o;
  return __uint_as_float(u);
}
__device__ __forceinline__ void atomic_add_f32(float* p, float v) {
#if defined(__HIP_DEVICE_COMPILE__)
  unsafeAtomicAdd(p, v);   // native global_atomic_add_f32
#else
  atomicAdd(p, v);
#endif
}

// ---------- utility ----------
__global__ void zero_u32_k(unsigned* p, size_t n) {
  size_t i = (size_t)blockIdx.x * 256 + threadIdx.x;
  size_t st = (size_t)gridDim.x * 256;
  for (; i < n; i += st) p[i] = 0u;
}

// Pack up to 4 column-concatenated weight matrices (each K x HC, row-major f32)
// into bf16 WMMA-B tiles. Tile (kt,nt) base = (nt*KTILES+kt)*512 elements; within a
// tile, lane l holds col = nt*16+(l&15), K = kt*32+(l>>4)*16 + p (p=0..15) contiguous.
__global__ void pack_b4_k(const float* __restrict__ W0, const float* __restrict__ W1,
                          const float* __restrict__ W2, const float* __restrict__ W3,
                          unsigned short* __restrict__ out, int HC, int Ncols) {
  int idx = blockIdx.x * 256 + threadIdx.x;
  if (idx >= KDIM * Ncols) return;
  int tile = idx >> 9, w = idx & 511;
  int lane = w >> 4, p = w & 15;
  int kt = tile % KTILES, nt = tile / KTILES;
  int col = nt * 16 + (lane & 15);
  int kk  = kt * 32 + (lane >> 4) * 16 + p;
  int mi = col / HC, lc = col - mi * HC;
  const float* W = (mi == 0) ? W0 : (mi == 1) ? W1 : (mi == 2) ? W2 : W3;
  out[idx] = f32_to_bf16(W[(size_t)kk * HC + lc]);
}

__global__ void bias_cat4_k(const float* __restrict__ b0, const float* __restrict__ b1,
                            const float* __restrict__ b2, const float* __restrict__ b3,
                            float* __restrict__ out, int HC, int Ncols) {
  int i = blockIdx.x * 256 + threadIdx.x;
  if (i >= Ncols) return;
  int mi = i / HC, lc = i - mi * HC;
  const float* b = (mi == 0) ? b0 : (mi == 1) ? b1 : (mi == 2) ? b2 : b3;
  out[i] = b[lc];
}

// ---------- WMMA GEMM: C[MxNcols] = A_bf16[MxK=128] @ Bpack (+bias)(+relu) ----------
// Block = 8 waves covering (8/WPB) M-tiles x WPB N-tiles. The contiguous packed-B
// panel for the block's N-tiles (<=32KB) is staged into LDS once by the Tensor Data
// Mover (TENSORcnt + s_wait_tensorcnt), then every wave feeds WMMA from LDS.
__global__ __launch_bounds__(256)
void gemm_bf16_k(const unsigned short* __restrict__ A, const unsigned short* __restrict__ Bp,
                 const float* __restrict__ bias, float* __restrict__ Cf,
                 unsigned short* __restrict__ Cb, int M, int Ncols, int relu) {
  __shared__ __align__(16) unsigned short bsh[8 * KTILES * 512];   // 32 KB max panel
  int lane = threadIdx.x & 31;
  int wid  = threadIdx.x >> 5;
  int Ntiles = Ncols >> 4;
  int Mtiles = (M + 15) >> 4;
  int WPB  = (Ntiles < 8) ? Ntiles : 8;     // N-tiles per block
  int nblk = Ntiles / WPB;
  int mpb  = 8 / WPB;                       // M-tiles per block
  int bm = blockIdx.x / nblk, bn = blockIdx.x % nblk;
  int ntbase = bn * WPB;
  int mt = bm * mpb + wid / WPB;
  int ln = wid % WPB;                       // local n-tile
  int panel8 = WPB * KTILES * 128;          // panel size in 8-byte units (<=4096)

#if __has_builtin(__builtin_amdgcn_tensor_load_to_lds)
  if (wid == 0) {
    unsigned long long ga = (unsigned long long)(Bp + (((size_t)ntbase * KTILES) << 9));
    unsigned ldsa = (unsigned)(unsigned long long)(void*)&bsh[0];
    v4u g0; v8i g1; v4i g2 = {0, 0, 0, 0}, g3 = {0, 0, 0, 0};
    v8i g4 = {0, 0, 0, 0, 0, 0, 0, 0};
    // D# group0: count=1 | lds_addr | global_addr | type=2 (bits 127:126)
    g0[0] = 1u;
    g0[1] = ldsa;
    g0[2] = (unsigned)ga;
    g0[3] = (unsigned)((ga >> 32) & 0x01ffffffu) | 0x80000000u;
    // D# group1: data_size=8B; 1-D tensor: dim0 = stride0 = tile_dim0 = panel8
    g1[0] = 0x00030000;                        // data_size=3 (8B), no multicast/pad
    g1[1] = (int)(((unsigned)panel8 & 0xffffu) << 16);          // tensor_dim0[15:0]
    g1[2] = (int)((((unsigned)panel8 >> 16) & 0xffffu) | (1u << 16)); // dim0 hi | tensor_dim1=1
    g1[3] = (int)(((unsigned)panel8 & 0xffffu) << 16);          // tile_dim0
    g1[4] = 0;                                                  // tile_dim1/2 unused
    g1[5] = panel8;                                             // tensor_dim0_stride lo
    g1[6] = 0;
    g1[7] = 0;
    __builtin_amdgcn_tensor_load_to_lds(g0, g1, g2, g3, g4, 0);
    __builtin_amdgcn_s_wait_tensorcnt(0);
  }
#else
  {  // fallback: cooperative copy global -> LDS
    const uint4* src = (const uint4*)(Bp + (((size_t)ntbase * KTILES) << 9));
    uint4* dstl = (uint4*)&bsh[0];
    int n16 = panel8 >> 1;                    // 16B chunks
    for (int i = threadIdx.x; i < n16; i += 256) dstl[i] = src[i];
  }
#endif
  __syncthreads();

  if (mt < Mtiles) {
    const unsigned short* Arow = A + (size_t)(mt * 16 + (lane & 15)) * KDIM + ((lane >> 4) * 16);
    const unsigned short* Bt   = &bsh[((size_t)ln * KTILES) << 9] + lane * 16;

    v8f c = {};
#pragma unroll
    for (int kt = 0; kt < KTILES; ++kt) {
      union { v16bf v; uint4 q[2]; } a, b;
      a.q[0] = *(const uint4*)(Arow + kt * 32);
      a.q[1] = *(const uint4*)(Arow + kt * 32 + 8);
      b.q[0] = *(const uint4*)(Bt);
      b.q[1] = *(const uint4*)(Bt + 8);
      Bt += 512;
      if (kt + 1 < KTILES)
        __builtin_prefetch((const void*)(Arow + (kt + 1) * 32), 0, 3);  // WGP-scope
      c = __builtin_amdgcn_wmma_f32_16x16x32_bf16(false, a.v, false, b.v, (short)0, c, false, false);
    }

    int n = lane & 15;
    int col = (ntbase + ln) * 16 + n;
    float bv = bias ? bias[col] : 0.0f;
    int rbase = mt * 16 + ((lane >> 4) << 3);
#pragma unroll
    for (int r = 0; r < 8; ++r) {
      int row = rbase + r;
      if (row < M) {
        float v = c[r] + bv;
        if (relu) v = fmaxf(v, 0.0f);
        if (Cf) Cf[(size_t)row * Ncols + col] = v;
        if (Cb) Cb[(size_t)row * Ncols + col] = f32_to_bf16(v);
      }
    }
  }
}

// ---------- small dense front-end kernels ----------
__global__ void graph_embed_k(const float* __restrict__ gf,
                              const float* __restrict__ W0, const float* __restrict__ b0,
                              const float* __restrict__ W1, const float* __restrict__ b1,
                              float* xf, unsigned short* xb) {
  int t = threadIdx.x;
  __shared__ float h1[128];
  float a = b0[t];
#pragma unroll
  for (int i = 0; i < 14; ++i) a += gf[i] * W0[i * 128 + t];
  h1[t] = fmaxf(a, 0.0f);
  __syncthreads();
  float o = b1[t];
  for (int i = 0; i < 128; ++i) o += h1[i] * W1[i * 128 + t];
  xf[t] = o;
  xb[t] = f32_to_bf16(o);
}

__global__ void node_proj_k(const float* __restrict__ nf, const float* __restrict__ W,
                            const float* __restrict__ b, float* xf, unsigned short* xb) {
  int row = blockIdx.x, t = threadIdx.x;
  const float* a = nf + (size_t)row * 8;
  float v = b[t];
#pragma unroll
  for (int i = 0; i < 8; ++i) v += a[i] * W[i * 128 + t];
  xf[(size_t)(row + 1) * 128 + t] = v;
  xb[(size_t)(row + 1) * 128 + t] = f32_to_bf16(v);
}

__global__ void edge_mlp1_k(const float* __restrict__ ea, const float* __restrict__ W,
                            const float* __restrict__ b, unsigned short* out) {
  int row = blockIdx.x, t = threadIdx.x;
  const float* a = ea + (size_t)row * 3;
  float v = b[t] + a[0] * W[t] + a[1] * W[128 + t] + a[2] * W[256 + t];
  out[(size_t)row * 128 + t] = f32_to_bf16(fmaxf(v, 0.0f));
}

// ---------- edge attention (segment softmax) ----------
__global__ void edge_score_k(const int* __restrict__ src, const int* __restrict__ dst,
                             const float* __restrict__ qkvs, const unsigned short* __restrict__ ee,
                             float* __restrict__ score, unsigned* __restrict__ mord, int HC) {
  int t = blockIdx.x * 256 + threadIdx.x;
  if (t >= NE * HEADS) return;
  int e = t >> 2, h = t & 3;
  int C = HC / HEADS;
  int s = src[e], d = dst[e];
  const float* q = qkvs + (size_t)d * 4 * HC + h * C;
  const float* k = qkvs + (size_t)s * 4 * HC + HC + h * C;
  const unsigned short* ev = ee + (size_t)e * HC + h * C;
  float acc = 0.0f;
  for (int c = 0; c < C; ++c) acc += q[c] * (k[c] + bf16_to_f32(ev[c]));
  acc *= rsqrtf((float)C);
  score[t] = acc;
  atomicMax(&mord[d * HEADS + h], ford(acc));
}

__global__ void edge_w_k(const int* __restrict__ dst, const float* __restrict__ score,
                         const unsigned* __restrict__ mord, float* __restrict__ wbuf,
                         float* __restrict__ denom) {
  int t = blockIdx.x * 256 + threadIdx.x;
  if (t >= NE * HEADS) return;
  int e = t >> 2, h = t & 3;
  int d = dst[e];
  float m = funord(mord[d * HEADS + h]);
  float w = __expf(score[t] - m);
  wbuf[t] = w;
  atomic_add_f32(&denom[d * HEADS + h], w);
}

__global__ void edge_msg_k(const int* __restrict__ src, const int* __restrict__ dst,
                           const float* __restrict__ qkvs, const unsigned short* __restrict__ ee,
                           const float* __restrict__ wbuf, const float* __restrict__ denom,
                           float* __restrict__ aggr, int HC) {
  int t = blockIdx.x * 256 + threadIdx.x;
  if (t >= NE * HEADS) return;
  int e = t >> 2, h = t & 3;
  int C = HC / HEADS;
  int s = src[e], d = dst[e];
  float al = wbuf[t] / denom[d * HEADS + h];
  const float* v = qkvs + (size_t)s * 4 * HC + 2 * HC + h * C;
  const unsigned short* ev = ee + (size_t)e * HC + h * C;
  float* o = aggr + (size_t)d * HC + h * C;
  for (int c = 0; c < C; ++c) atomic_add_f32(&o[c], (v[c] + bf16_to_f32(ev[c])) * al);
}

// ---------- residual + layernorm (+relu), in-place on x ----------
__global__ void post_ln_k(const float* __restrict__ aggr, const float* __restrict__ qkvs,
                          const float* __restrict__ gam, const float* __restrict__ bet,
                          float* xf, unsigned short* xb, int HC, int addres, int dorelu) {
  int row = blockIdx.x, c = threadIdx.x;   // blockDim == HC
  __shared__ float red[128];
  float v = aggr[(size_t)row * HC + c] + qkvs[(size_t)row * 4 * HC + 3 * HC + c];
  if (addres) v += xf[(size_t)row * HC + c];
  red[c] = v; __syncthreads();
  for (int st = HC >> 1; st > 0; st >>= 1) { if (c < st) red[c] += red[c + st]; __syncthreads(); }
  float mu = red[0] / HC; __syncthreads();
  float d = v - mu;
  red[c] = d * d; __syncthreads();
  for (int st = HC >> 1; st > 0; st >>= 1) { if (c < st) red[c] += red[c + st]; __syncthreads(); }
  float var = red[0] / HC;
  float y = d * rsqrtf(var + 1e-5f) * gam[c] + bet[c];
  if (dorelu) y = fmaxf(y, 0.0f);
  xf[(size_t)row * HC + c] = y;
  xb[(size_t)row * HC + c] = f32_to_bf16(y);
}

// h = concat(nodes, broadcast(gtok)) as bf16; x stride is 64 after last conv layer
__global__ void build_h_k(const float* __restrict__ xf, unsigned short* hb) {
  int row = blockIdx.x, c = threadIdx.x;
  float v = 0.0f;
  if (row < N_NODES) v = (c < 64) ? xf[(size_t)(row + 1) * 64 + c] : xf[c - 64];
  hb[(size_t)row * 128 + c] = f32_to_bf16(v);
}

// ---------- output heads ----------
__device__ __forceinline__ float blk_dot(float partial, float* red, int t) {
  red[t] = partial; __syncthreads();
  for (int st = 64; st > 0; st >>= 1) { if (t < st) red[t] += red[t + st]; __syncthreads(); }
  float s = red[0]; __syncthreads();
  return s;
}
__global__ void heads_k(const float* __restrict__ h2,
                        const float* Wa, const float* ba, const float* Wg, const float* bg,
                        const float* Wal, const float* bal, const float* Wn, const float* bn,
                        const float* Wt, const float* bt, float* __restrict__ out) {
  int row = blockIdx.x, t = threadIdx.x;
  __shared__ float red[128];
  float hv = h2[(size_t)row * 128 + t];
  float l0 = blk_dot(hv * Wa[t * 2 + 0], red, t) + ba[0];
  float l1 = blk_dot(hv * Wa[t * 2 + 1], red, t) + ba[1];
  float gm = blk_dot(hv * Wg[t],  red, t) + bg[0];
  float al = blk_dot(hv * Wal[t], red, t) + bal[0];
  float nz = blk_dot(hv * Wn[t],  red, t) + bn[0];
  float th = blk_dot(hv * Wt[t],  red, t) + bt[0];
  if (t == 0) {
    float* o = out + (size_t)row * 6;
    o[0] = l0;
    o[1] = l1;
    o[2] = 5.0f / (1.0f + __expf(-gm));
    o[3] = 2.0f / (1.0f + __expf(-al));
    o[4] = 1.0f / (1.0f + __expf(-nz));
    o[5] = 3.14159265358979323846f * tanhf(th);
  }
}

// ---------- host orchestration ----------
extern "C" void kernel_launch(void* const* d_in, const int* in_sizes, int n_in,
                              void* d_out, int out_size, void* d_ws, size_t ws_size,
                              hipStream_t stream) {
  (void)in_sizes; (void)n_in; (void)out_size; (void)ws_size;
  auto F = [&](int i) { return (const float*)d_in[i]; };
  const float* GF = F(0);
  const float* NF = F(1);
  const float* EA = F(2);
  const int*   EI = (const int*)d_in[3];
  const int* SRC = EI;
  const int* DST = EI + NE;
  // params, flattened in setup_inputs() insertion order
  const float *gm0W = F(4), *gm0b = F(5), *gm1W = F(6), *gm1b = F(7);
  const float *npW = F(8), *npb = F(9);
  const float *em0W = F(10), *em0b = F(11), *em1W = F(12), *em1b = F(13);
  // convs at 14 + 9*i : qW qb kW kb vW vb eW skW skb ; norms at 50 + 2*i : g b
  const float *nm0W = F(58), *nm0b = F(59), *nm1W = F(60), *nm1b = F(61);
  const float *aW = F(62), *ab = F(63), *gW = F(64), *gb = F(65);
  const float *alW = F(66), *alb = F(67), *noW = F(68), *nob = F(69);
  const float *thW = F(70), *thb = F(71);

  char* wsp = (char*)d_ws;
  auto alloc = [&](size_t bytes) -> void* {
    void* r = (void*)wsp; wsp += (bytes + 255) & ~(size_t)255; return r;
  };
  float*          xf     = (float*)alloc((size_t)NPAD * 128 * 4);
  unsigned short* xb     = (unsigned short*)alloc((size_t)NPAD * 128 * 2);
  unsigned short* ebuf   = (unsigned short*)alloc((size_t)NE * 128 * 2);   // edge emb (bf16)
  unsigned short* eebuf  = (unsigned short*)alloc((size_t)NE * 128 * 2);   // per-layer e-proj / scratch
  float*          qkvs   = (float*)alloc((size_t)NPAD * 512 * 4);
  float*          score  = (float*)alloc((size_t)NE * 4 * 4);
  float*          wbuf   = (float*)alloc((size_t)NE * 4 * 4);
  unsigned*       mord   = (unsigned*)alloc((size_t)NPAD * 4 * 4);
  float*          denom  = (float*)alloc((size_t)NPAD * 4 * 4);
  float*          aggr   = (float*)alloc((size_t)NPAD * 128 * 4);
  unsigned short* wpack  = (unsigned short*)alloc((size_t)128 * 512 * 2);
  float*          biascat= (float*)alloc(512 * 4);
  unsigned short* hcat   = (unsigned short*)alloc((size_t)NPAD * 128 * 2);
  unsigned short* h1b    = (unsigned short*)alloc((size_t)N_NODES * 128 * 2);
  float*          h2f    = (float*)alloc((size_t)N_NODES * 128 * 4);

  auto cdiv = [](long a, long b) { return (int)((a + b - 1) / b); };
  auto gemm_grid = [&](int M, int Ncols) {
    int Ntiles = Ncols >> 4;
    int Mtiles = (M + 15) >> 4;
    int WPB = (Ntiles < 8) ? Ntiles : 8;
    int nblk = Ntiles / WPB;
    int mpb = 8 / WPB;
    return cdiv(Mtiles, mpb) * nblk;
  };

  // zero the bf16 A-matrix pad rows (read by WMMA tiles, never written)
  zero_u32_k<<<4, 256, 0, stream>>>((unsigned*)(xb + (size_t)NV * 128), (size_t)(NPAD - NV) * 128 / 2);

  // front end
  graph_embed_k<<<1, 128, 0, stream>>>(GF, gm0W, gm0b, gm1W, gm1b, xf, xb);
  node_proj_k<<<N_NODES, 128, 0, stream>>>(NF, npW, npb, xf, xb);
  edge_mlp1_k<<<NE, 128, 0, stream>>>(EA, em0W, em0b, eebuf);
  pack_b4_k<<<cdiv(128 * 128, 256), 256, 0, stream>>>(em1W, em1W, em1W, em1W, wpack, 128, 128);
  bias_cat4_k<<<1, 256, 0, stream>>>(em1b, em1b, em1b, em1b, biascat, 128, 128);
  gemm_bf16_k<<<gemm_grid(NE, 128), 256, 0, stream>>>(eebuf, wpack, biascat,
                                                      nullptr, ebuf, NE, 128, 0);

  for (int i = 0; i < 4; ++i) {
    int HC = (i == 3) ? 64 : 128;
    int Ncols = 4 * HC;
    int base = 14 + i * 9;
    const float *qWp = F(base + 0), *qbp = F(base + 1);
    const float *kWp = F(base + 2), *kbp = F(base + 3);
    const float *vWp = F(base + 4), *vbp = F(base + 5);
    const float *eWp = F(base + 6);
    const float *sWp = F(base + 7), *sbp = F(base + 8);
    const float *ng = F(50 + i * 2), *nb = F(51 + i * 2);

    zero_u32_k<<<cdiv((long)NPAD * 4, 256), 256, 0, stream>>>(mord, (size_t)NPAD * 4);
    zero_u32_k<<<cdiv((long)NPAD * 4, 256), 256, 0, stream>>>((unsigned*)denom, (size_t)NPAD * 4);
    zero_u32_k<<<4096, 256, 0, stream>>>((unsigned*)aggr, (size_t)NPAD * 128);

    // q|k|v|skip fused projection
    pack_b4_k<<<cdiv(128L * Ncols, 256), 256, 0, stream>>>(qWp, kWp, vWp, sWp, wpack, HC, Ncols);
    bias_cat4_k<<<cdiv(Ncols, 256), 256, 0, stream>>>(qbp, kbp, vbp, sbp, biascat, HC, Ncols);
    gemm_bf16_k<<<gemm_grid(NV, Ncols), 256, 0, stream>>>(xb, wpack, biascat,
                                                          qkvs, nullptr, NV, Ncols, 0);

    // e-projection (no bias), bf16 output
    pack_b4_k<<<cdiv(128L * HC, 256), 256, 0, stream>>>(eWp, eWp, eWp, eWp, wpack, HC, HC);
    gemm_bf16_k<<<gemm_grid(NE, HC), 256, 0, stream>>>(ebuf, wpack, nullptr,
                                                       nullptr, eebuf, NE, HC, 0);

    int eg = cdiv((long)NE * HEADS, 256);
    edge_score_k<<<eg, 256, 0, stream>>>(SRC, DST, qkvs, eebuf, score, mord, HC);
    edge_w_k<<<eg, 256, 0, stream>>>(DST, score, mord, wbuf, denom);
    edge_msg_k<<<eg, 256, 0, stream>>>(SRC, DST, qkvs, eebuf, wbuf, denom, aggr, HC);

    post_ln_k<<<NV, HC, 0, stream>>>(aggr, qkvs, ng, nb, xf, xb, HC,
                                     (i < 3) ? 1 : 0, (i < 3) ? 1 : 0);
  }

  // policy MLP + heads
  build_h_k<<<NPAD, 128, 0, stream>>>(xf, hcat);
  pack_b4_k<<<cdiv(128 * 128, 256), 256, 0, stream>>>(nm0W, nm0W, nm0W, nm0W, wpack, 128, 128);
  bias_cat4_k<<<1, 256, 0, stream>>>(nm0b, nm0b, nm0b, nm0b, biascat, 128, 128);
  gemm_bf16_k<<<gemm_grid(N_NODES, 128), 256, 0, stream>>>(hcat, wpack, biascat,
                                                           nullptr, h1b, N_NODES, 128, 1);
  pack_b4_k<<<cdiv(128 * 128, 256), 256, 0, stream>>>(nm1W, nm1W, nm1W, nm1W, wpack, 128, 128);
  bias_cat4_k<<<1, 256, 0, stream>>>(nm1b, nm1b, nm1b, nm1b, biascat, 128, 128);
  gemm_bf16_k<<<gemm_grid(N_NODES, 128), 256, 0, stream>>>(h1b, wpack, biascat,
                                                           h2f, nullptr, N_NODES, 128, 1);
  heads_k<<<N_NODES, 128, 0, stream>>>(h2f, aW, ab, gW, gb, alW, alb, noW, nob, thW, thb,
                                       (float*)d_out);
}
